// Attention_20237885899303
// MI455X (gfx1250) — matrix operand order repaired and checked by reading, not compiled, not run on hardware
//
#include <hip/hip_runtime.h>
#include <hip/hip_bf16.h>

typedef __bf16 bf16;
typedef __attribute__((ext_vector_type(16))) __bf16 v16bf;
typedef __attribute__((ext_vector_type(8)))  __bf16 v8bf;
typedef __attribute__((ext_vector_type(8)))  float  v8f;

#define DIMX 512
#define HEADS 8
#define DIM_HEAD 64
#define INNER 512
#define BATCH 8
#define NSEQ 256
#define NUNIQ 961          // (2*16-1)^2
#define QSCALE 0.125f      // 64^-0.5
#define LN_EPS 1e-5f

// ---------------------------------------------------------------- WMMA helpers
__device__ __forceinline__ v8f wmma_bf16(v16bf a, v16bf b, v8f c) {
  return __builtin_amdgcn_wmma_f32_16x16x32_bf16(
      /*neg_a=*/false, a, /*neg_b=*/false, b,
      /*c_mod=*/(short)0, c, /*reuse_a=*/false, /*reuse_b=*/false);
}

// A 16x32 bf16 row-major (lda elems). Lane slice is two contiguous 16B runs.
__device__ __forceinline__ v16bf load_frag_a(const bf16* __restrict__ A, int lda) {
  int lane = threadIdx.x & 31;
  const bf16* p = A + (size_t)(lane & 15) * lda + ((lane >> 4) << 4);
  v8bf lo = *(const v8bf*)p;
  v8bf hi = *(const v8bf*)(p + 8);
  return __builtin_shufflevector(lo, hi, 0,1,2,3,4,5,6,7,8,9,10,11,12,13,14,15);
}

// A fragment from f32 source (converted on load), two aligned 32B runs.
__device__ __forceinline__ v16bf load_frag_a_f32(const float* __restrict__ A, int lda) {
  int lane = threadIdx.x & 31;
  const float* p = A + (size_t)(lane & 15) * lda + ((lane >> 4) << 4);
  v8f lo = *(const v8f*)p;
  v8f hi = *(const v8f*)(p + 8);
  v16bf f;
#pragma unroll
  for (int i = 0; i < 8; ++i) { f[i] = (bf16)lo[i]; f[i + 8] = (bf16)hi[i]; }
  return f;
}

// B 32x16 from transposed storage (N x K row-major): one aligned 32B run per lane.
__device__ __forceinline__ v16bf load_frag_bt(const bf16* __restrict__ Bt, int ldbt) {
  int lane = threadIdx.x & 31;
  const bf16* p = Bt + (size_t)(lane & 15) * ldbt + ((lane >> 4) << 4);
  return *(const v16bf*)p;
}

// B 32x16 from packed fragment-major storage: tile t, lane slice contiguous.
// packed[(t*32 + lane)*16 + i] == B[kt*32 + (lane>>4)*16 + i][nt*16 + (lane&15)]
__device__ __forceinline__ v16bf load_frag_b_packed(const bf16* __restrict__ P, int tile) {
  int lane = threadIdx.x & 31;
  return *(const v16bf*)(P + ((size_t)tile * 32 + lane) * 16);
}

// C/D 16x16 f32: VGPR j -> M = j + 8*(lane>=16), N = lane&15.
__device__ __forceinline__ int c_row(int j) {
  return ((threadIdx.x & 31) >> 4) * 8 + j;
}
__device__ __forceinline__ int c_col() { return threadIdx.x & 15; }

// Packed-B index for element (k, n) of a K x N matrix with ntiles = N/16.
__device__ __forceinline__ size_t pack_b_index(int k, int n, int ntiles) {
  int kt = k >> 5, kw = k & 31;
  int lane = ((kw >> 4) << 4) + (n & 15);
  return ((size_t)(kt * ntiles + (n >> 4)) * 32 + lane) * 16 + (kw & 15);
}

// ---------------------------------------------------------------- kernels
// Convert + repack weights: w_qkv / w_out into fragment-major bf16; uniq plain bf16.
__global__ void prep_weights(const float* __restrict__ w_qkv,
                             const float* __restrict__ uniq,
                             const float* __restrict__ w_out,
                             bf16* __restrict__ w_qkv_p,
                             bf16* __restrict__ uniq_b,
                             bf16* __restrict__ w_out_p) {
  int i = blockIdx.x * blockDim.x + threadIdx.x;
  const int n1 = DIMX * 3 * INNER;   // 786432
  const int n2 = NUNIQ * INNER;      // 492032
  const int n3 = INNER * DIMX;       // 262144
  if (i < n1) {
    int k = i / (3 * INNER), n = i % (3 * INNER);
    w_qkv_p[pack_b_index(k, n, 96)] = (bf16)w_qkv[i];
  }
  if (i < n2) uniq_b[i] = (bf16)uniq[i];
  if (i < n3) {
    int k = i / DIMX, n = i % DIMX;
    w_out_p[pack_b_index(k, n, 32)] = (bf16)w_out[i];
  }
}

__global__ void layernorm_kernel(const float* __restrict__ x,
                                 const float* __restrict__ g,
                                 const float* __restrict__ beta,
                                 bf16* __restrict__ xn) {
  int row = blockIdx.x;                       // 0..2047
  const float* xr = x + (size_t)row * DIMX;
  int t = threadIdx.x;                        // 256 threads, 2 elems each
  float v0 = xr[t], v1 = xr[t + 256];
  __shared__ float red[8];
  float s = v0 + v1;
#pragma unroll
  for (int o = 16; o > 0; o >>= 1) s += __shfl_xor(s, o, 32);
  if ((t & 31) == 0) red[t >> 5] = s;
  __syncthreads();
  float mean = 0.f;
#pragma unroll
  for (int i = 0; i < 8; ++i) mean += red[i];
  mean *= (1.0f / (float)DIMX);
  __syncthreads();
  float d0 = v0 - mean, d1 = v1 - mean;
  float q = d0 * d0 + d1 * d1;
#pragma unroll
  for (int o = 16; o > 0; o >>= 1) q += __shfl_xor(q, o, 32);
  if ((t & 31) == 0) red[t >> 5] = q;
  __syncthreads();
  float var = 0.f;
#pragma unroll
  for (int i = 0; i < 8; ++i) var += red[i];
  var *= (1.0f / (float)DIMX);
  float rstd = rsqrtf(var + LN_EPS);
  xn[(size_t)row * DIMX + t]       = (bf16)(d0 * rstd * g[t] + beta[t]);
  xn[(size_t)row * DIMX + t + 256] = (bf16)(d1 * rstd * g[t + 256] + beta[t + 256]);
}

// C(2048x1536) = xn @ w_qkv. One wave -> 16x64 strip (4 acc, A reused 4x).
// q,k stored [b,h,n,d] bf16; v stored packed fragment-major per (b,h).
__global__ void qkv_gemm(const bf16* __restrict__ xn, const bf16* __restrict__ wp,
                         bf16* __restrict__ q, bf16* __restrict__ k,
                         bf16* __restrict__ vpk) {
  int gw = (blockIdx.x * blockDim.x + threadIdx.x) >> 5;   // 0..3071
  int mt = gw / 24;           // 128 row tiles
  int ng = gw % 24;           // 24 groups of 4 col tiles (64 cols)
  const bf16* A0 = xn + (size_t)mt * 16 * DIMX;
  v8f acc[4] = {{}, {}, {}, {}};
#pragma unroll 2
  for (int kk = 0; kk < DIMX; kk += 32) {
    __builtin_prefetch(A0 + kk + 128, 0, 1);
    v16bf a = load_frag_a(A0 + kk, DIMX);
    int kt = kk >> 5;
#pragma unroll
    for (int u = 0; u < 4; ++u)
      acc[u] = wmma_bf16(a, load_frag_b_packed(wp, kt * 96 + ng * 4 + u), acc[u]);
  }
#pragma unroll
  for (int u = 0; u < 4; ++u) {
    int ncol = (ng * 4 + u) * 16 + c_col();
    int part = ncol / INNER;
    int inner = ncol % INNER;
    int hh = inner / DIM_HEAD, dd = inner % DIM_HEAD;
#pragma unroll
    for (int j = 0; j < 8; ++j) {
      int mrow = mt * 16 + c_row(j);
      int b_ = mrow >> 8, n_ = mrow & 255;
      float val = acc[u][j];
      if (part == 0) {
        q[(((size_t)b_ * HEADS + hh) * NSEQ + n_) * DIM_HEAD + dd] = (bf16)(val * QSCALE);
      } else if (part == 1) {
        k[(((size_t)b_ * HEADS + hh) * NSEQ + n_) * DIM_HEAD + dd] = (bf16)val;
      } else {
        // v packed per (b,h): K = m (seq), N = d, ntiles = 4
        int kt2 = n_ >> 5, kw = n_ & 31;
        int lanep = ((kw >> 4) << 4) + (dd & 15);
        size_t dst = ((size_t)(b_ * HEADS + hh) * 32 + (size_t)kt2 * 4 + (dd >> 4)) * 512
                   + (size_t)lanep * 16 + (kw & 15);
        vpk[dst] = (bf16)val;
      }
    }
  }
}

// Per (b, h, 16-row n-tile): S = q_scaled @ k^T (16x256), softmax, bf16 attn.
__global__ void attn_softmax(const bf16* __restrict__ q, const bf16* __restrict__ k,
                             bf16* __restrict__ attn) {
  int bid = blockIdx.x;            // b*128 + h*16 + ntile
  int ntile = bid & 15;
  int h = (bid >> 4) & 7;
  int b = bid >> 7;
  const bf16* qb = q + (((size_t)b * HEADS + h) * NSEQ + ntile * 16) * DIM_HEAD;
  const bf16* kb = k + ((size_t)b * HEADS + h) * NSEQ * DIM_HEAD;
  __shared__ float S[16][258];
  __shared__ float red2[16][17];
  int wid = threadIdx.x >> 5;
#pragma unroll
  for (int c = 0; c < 2; ++c) {
    int mtile = wid * 2 + c;       // 8 waves x 2 = 16 column tiles
    v8f acc = {};
#pragma unroll
    for (int kk = 0; kk < DIM_HEAD; kk += 32) {
      v16bf a  = load_frag_a(qb + kk, DIM_HEAD);
      v16bf bt = load_frag_bt(kb + (size_t)mtile * 16 * DIM_HEAD + kk, DIM_HEAD);
      acc = wmma_bf16(a, bt, acc);
    }
#pragma unroll
    for (int j = 0; j < 8; ++j)
      S[c_row(j)][mtile * 16 + c_col()] = acc[j];
  }
  __syncthreads();
  int r = threadIdx.x >> 4;        // 16 rows
  int c16 = threadIdx.x & 15;      // 16 threads per row, 16 cols each
  float lmax = -1e30f;
#pragma unroll
  for (int i = 0; i < 16; ++i) lmax = fmaxf(lmax, S[r][c16 * 16 + i]);
  red2[r][c16] = lmax;
  __syncthreads();
  float rmax = red2[r][0];
#pragma unroll
  for (int i = 1; i < 16; ++i) rmax = fmaxf(rmax, red2[r][i]);
  float ev[16];
  float lsum = 0.f;
#pragma unroll
  for (int i = 0; i < 16; ++i) { ev[i] = __expf(S[r][c16 * 16 + i] - rmax); lsum += ev[i]; }
  __syncthreads();
  red2[r][c16] = lsum;
  __syncthreads();
  float rsum = 0.f;
#pragma unroll
  for (int i = 0; i < 16; ++i) rsum += red2[r][i];
  float inv = 1.0f / rsum;
  int n = ntile * 16 + r;
  v16bf orow;
#pragma unroll
  for (int i = 0; i < 16; ++i) orow[i] = (bf16)(ev[i] * inv);
  *(v16bf*)(attn + (((size_t)b * HEADS + h) * NSEQ + n) * NSEQ + c16 * 16) = orow;
}

// ctx[b,n,h,:] = attn[b,h,n,:] @ v[b,h,:,:]. One wave -> 16(n) x 64(d), 4 acc.
__global__ void attn_v(const bf16* __restrict__ attn, const bf16* __restrict__ vpk,
                       float* __restrict__ ctx) {
  int gw = (blockIdx.x * blockDim.x + threadIdx.x) >> 5;  // 0..1023
  int nt = gw & 15;
  int h  = (gw >> 4) & 7;
  int b  = gw >> 7;
  const bf16* A0 = attn + (((size_t)b * HEADS + h) * NSEQ + nt * 16) * NSEQ;
  const bf16* Vp = vpk + (size_t)(b * HEADS + h) * 32 * 512;
  v8f acc[4] = {{}, {}, {}, {}};
#pragma unroll 2
  for (int kk = 0; kk < NSEQ; kk += 32) {
    __builtin_prefetch(A0 + kk + 128, 0, 1);
    v16bf a = load_frag_a(A0 + kk, NSEQ);
    int kt = kk >> 5;
#pragma unroll
    for (int u = 0; u < 4; ++u)
      acc[u] = wmma_bf16(a, load_frag_b_packed(Vp, kt * 4 + u), acc[u]);
  }
#pragma unroll
  for (int u = 0; u < 4; ++u) {
    int d = u * 16 + c_col();
#pragma unroll
    for (int j = 0; j < 8; ++j) {
      int n = nt * 16 + c_row(j);
      ctx[((size_t)b * NSEQ + n) * INNER + h * DIM_HEAD + d] = acc[u][j];
    }
  }
}

// ctx[b,n,h,d] += sum_m attn[b,h,n,m] * unique[idx_map[n,m], h*64+d]
// One wave per (h,n): A = attn rows b=0..7 (padded to 16), B gathered via idx_map.
__global__ void attn_rpos(const bf16* __restrict__ attn, const bf16* __restrict__ uniq_b,
                          const int* __restrict__ idx_map, float* __restrict__ ctx) {
  int h = blockIdx.x >> 8;
  int n = blockIdx.x & 255;
  __shared__ int idx[256];
  for (int i = threadIdx.x; i < 256; i += 32) idx[i] = idx_map[n * 256 + i];
  __syncthreads();
  int lane = threadIdx.x & 31;
  int mrow = lane & 15;          // padded-batch row for A fragment
  int rb   = mrow & 7;           // clamp so loads stay in-bounds
  int kbA  = (lane >> 4) << 4;
  int nc   = lane & 15;
  int kbB  = (lane >> 4) << 4;
  v8f acc[4] = {{}, {}, {}, {}};
  for (int kk = 0; kk < NSEQ; kk += 32) {
    const bf16* arow = attn + (((size_t)rb * HEADS + h) * NSEQ + n) * NSEQ + kk + kbA;
    v8bf t0 = *(const v8bf*)arow;
    v8bf t1 = *(const v8bf*)(arow + 8);
    v16bf a = __builtin_shufflevector(t0, t1, 0,1,2,3,4,5,6,7,8,9,10,11,12,13,14,15);
#pragma unroll
    for (int i = 0; i < 16; ++i) a[i] = (mrow < 8) ? a[i] : (bf16)0.0f;
#pragma unroll
    for (int dt = 0; dt < 4; ++dt) {
      v16bf bfrag;
#pragma unroll
      for (int i = 0; i < 16; ++i) {
        int km = kk + kbB + i;
        bfrag[i] = uniq_b[(size_t)idx[km] * INNER + h * DIM_HEAD + dt * 16 + nc];
      }
      acc[dt] = wmma_bf16(a, bfrag, acc[dt]);
    }
  }
#pragma unroll
  for (int dt = 0; dt < 4; ++dt) {
    int d = dt * 16 + c_col();
#pragma unroll
    for (int j = 0; j < 8; ++j) {
      int r = c_row(j);
      if (r < 8) {
        size_t off = ((size_t)r * NSEQ + n) * INNER + h * DIM_HEAD + d;
        ctx[off] += acc[dt][j];    // disjoint (h,n) blocks: no atomics needed
      }
    }
  }
}

// out(2048x512) = ctx @ w_out + b_out. One wave -> 16 x 64 strip (4 acc).
__global__ void out_proj(const float* __restrict__ ctx, const bf16* __restrict__ wp,
                         const float* __restrict__ b_out, float* __restrict__ out) {
  int gw = (blockIdx.x * blockDim.x + threadIdx.x) >> 5;  // 0..1023
  int ng = gw & 7;            // 8 groups of 4 col tiles
  int mt = gw >> 3;           // 128 row tiles
  const float* A0 = ctx + (size_t)mt * 16 * INNER;
  v8f acc[4] = {{}, {}, {}, {}};
#pragma unroll 2
  for (int kk = 0; kk < INNER; kk += 32) {
    __builtin_prefetch(A0 + kk + 128, 0, 1);
    v16bf a = load_frag_a_f32(A0 + kk, INNER);
    int kt = kk >> 5;
#pragma unroll
    for (int u = 0; u < 4; ++u)
      acc[u] = wmma_bf16(a, load_frag_b_packed(wp, kt * 32 + ng * 4 + u), acc[u]);
  }
#pragma unroll
  for (int u = 0; u < 4; ++u) {
    int col = (ng * 4 + u) * 16 + c_col();
    float bias = b_out[col];
#pragma unroll
    for (int j = 0; j < 8; ++j) {
      int row = mt * 16 + c_row(j);
      out[(size_t)row * DIMX + col] = acc[u][j] + bias;
    }
  }
}

// ---------------------------------------------------------------- launch
extern "C" void kernel_launch(void* const* d_in, const int* in_sizes, int n_in,
                              void* d_out, int out_size, void* d_ws, size_t ws_size,
                              hipStream_t stream) {
  (void)in_sizes; (void)n_in; (void)out_size; (void)ws_size;
  const float* x      = (const float*)d_in[0];
  const float* ln_g   = (const float*)d_in[1];
  const float* ln_b   = (const float*)d_in[2];
  const float* w_qkv  = (const float*)d_in[3];
  const float* uniq   = (const float*)d_in[4];
  const float* w_out  = (const float*)d_in[5];
  const float* b_out  = (const float*)d_in[6];
  const int*   idxmap = (const int*)d_in[7];
  float* out = (float*)d_out;

  unsigned char* ws = (unsigned char*)d_ws;
  size_t off = 0;
  auto carve = [&](size_t bytes) -> void* {
    void* p = ws + off;
    off += (bytes + 255) & ~(size_t)255;
    return p;
  };
  bf16*  wqkv_p = (bf16*)carve((size_t)DIMX * 3 * INNER * 2);          // 1.5 MB packed
  bf16*  uniq_b = (bf16*)carve((size_t)NUNIQ * INNER * 2);             // ~1 MB
  bf16*  wout_p = (bf16*)carve((size_t)INNER * DIMX * 2);              // 0.5 MB packed
  bf16*  xn_b   = (bf16*)carve((size_t)BATCH * NSEQ * DIMX * 2);       // 2 MB
  bf16*  q_b    = (bf16*)carve((size_t)BATCH * HEADS * NSEQ * DIM_HEAD * 2);
  bf16*  k_b    = (bf16*)carve((size_t)BATCH * HEADS * NSEQ * DIM_HEAD * 2);
  bf16*  v_pk   = (bf16*)carve((size_t)BATCH * HEADS * NSEQ * DIM_HEAD * 2); // packed
  bf16*  attn_b = (bf16*)carve((size_t)BATCH * HEADS * NSEQ * NSEQ * 2);     // 8 MB
  float* ctx    = (float*)carve((size_t)BATCH * NSEQ * INNER * 4);           // 4 MB

  prep_weights    <<<3072, 256, 0, stream>>>(w_qkv, uniq, w_out, wqkv_p, uniq_b, wout_p);
  layernorm_kernel<<<BATCH * NSEQ, 256, 0, stream>>>(x, ln_g, ln_b, xn_b);
  qkv_gemm        <<<(128 * 24) / 8, 256, 0, stream>>>(xn_b, wqkv_p, q_b, k_b, v_pk);
  attn_softmax    <<<BATCH * HEADS * 16, 256, 0, stream>>>(q_b, k_b, attn_b);
  attn_v          <<<(BATCH * HEADS * 16) / 8, 256, 0, stream>>>(attn_b, v_pk, ctx);
  attn_rpos       <<<HEADS * NSEQ, 32, 0, stream>>>(attn_b, uniq_b, idxmap, ctx);
  out_proj        <<<(128 * 8) / 8, 256, 0, stream>>>(ctx, wout_p, b_out, out);
}
